// ViTMoEAttention_24618752540911
// MI455X (gfx1250) — matrix operand order repaired and checked by reading, not compiled, not run on hardware
//
#include <hip/hip_runtime.h>
#include <hip/hip_fp16.h>

#define DEVINL __device__ __forceinline__

typedef __attribute__((ext_vector_type(16))) _Float16 v16h;
typedef __attribute__((ext_vector_type(8)))  float    v8f;

constexpr int NB  = 64;    // batch
constexpr int SEQ = 257;   // sequence length
constexpr int TP  = 288;   // padded seq = 18*16 = 9*32
constexpr int DIM = 1024;
constexpr int NH  = 16;
constexpr int DH  = 64;
constexpr int NE  = 8;
constexpr int RK  = 16;
constexpr int KR  = 32;    // top_k * rank
constexpr int NT16 = 18;   // TP / 16
constexpr int NT32 = 9;    // TP / 32
constexpr int SP  = 288;   // s-dim of transposed V (== TP, 9 chunks of 32)
constexpr float ATT_SCALE = 0.125f;   // 64^-0.5

union H8 { uint4 u; _Float16 h[8]; };

DEVINL v8f vzero() {
  v8f z;
#pragma unroll
  for (int i = 0; i < 8; ++i) z[i] = 0.f;
  return z;
}

// A-fragment (16x32 f16): lane holds row M=lane&15; chunk0 = 8 halves at p,
// chunk1 = 8 halves at p+16 (caller bakes in kk + hb*8).
DEVINL v16h ld_afrag(const _Float16* p) {
  H8 a, b;
  a.u = *(const uint4*)p;
  b.u = *(const uint4*)(p + 16);
  v16h r;
#pragma unroll
  for (int i = 0; i < 8; ++i) { r[i] = a.h[i]; r[i + 8] = b.h[i]; }
  return r;
}

// B-fragment (32x16 f16): lane holds col N=lane&15; chunk0 = 8 halves at p,
// chunk1 = 8 halves at p+16 (caller bakes in kk + hb*8).
DEVINL v16h ld_bfrag(const _Float16* p) {
  H8 a, b;
  a.u = *(const uint4*)p;
  b.u = *(const uint4*)(p + 16);
  v16h r;
#pragma unroll
  for (int i = 0; i < 8; ++i) { r[i] = a.h[i]; r[i + 8] = b.h[i]; }
  return r;
}

DEVINL v8f wmma16(v16h a, v16h b, v8f c) {
  return __builtin_amdgcn_wmma_f32_16x16x32_f16(false, a, false, b, (short)0, c,
                                                false, false);
}

// ---------------- conversion kernels ----------------

__global__ void k_cvt_x(const float* __restrict__ x, _Float16* __restrict__ out) {
  size_t i = (size_t)blockIdx.x * 256 + threadIdx.x;
  if (i >= (size_t)NB * TP * DIM) return;
  int d = (int)(i % DIM);
  int t = (int)((i / DIM) % TP);
  int b = (int)(i / ((size_t)DIM * TP));
  float v = (t < SEQ) ? x[((size_t)b * SEQ + t) * DIM + d] : 0.f;
  out[i] = (_Float16)v;
}

__global__ void k_cvt(const float* __restrict__ src, _Float16* __restrict__ dst, int n) {
  int i = blockIdx.x * 256 + threadIdx.x;
  if (i < n) dst[i] = (_Float16)src[i];
}

// Ucomb^T[b][o][kr] = U[e_k][o][r],  kr = k*16 + r
__global__ void k_ucombt(const float* __restrict__ U, const int* __restrict__ idx,
                         _Float16* __restrict__ out) {
  int i = blockIdx.x * 256 + threadIdx.x;   // [0, NB*DIM)
  if (i >= NB * DIM) return;
  int b = i >> 10, o = i & 1023;
  _Float16* dst = out + (size_t)i * KR;
#pragma unroll
  for (int k = 0; k < 2; ++k) {
    int e = idx[b * 2 + k];
    const float* src = U + ((size_t)e * DIM + o) * RK;
#pragma unroll
    for (int r = 0; r < RK; ++r) dst[k * 16 + r] = (_Float16)src[r];
  }
}

// ---------------- xv = (X @ Vexp^T) * (S*gate)  -> f16 [NB][TP][32] ----------

__global__ void k_xv(const _Float16* __restrict__ xin, const _Float16* __restrict__ vexp,
                     const float* __restrict__ sexp, const float* __restrict__ gates,
                     const int* __restrict__ idx, _Float16* __restrict__ xvs) {
  int w    = blockIdx.x * 8 + (threadIdx.x >> 5);
  int lane = threadIdx.x & 31;
  int b = w / NT16, mt = w % NT16;
  int ln = lane & 15, hb = lane >> 4;

  const _Float16* arow = xin + ((size_t)(b * TP + mt * 16 + ln)) * DIM;
  int e0 = idx[b * 2 + 0], e1 = idx[b * 2 + 1];
  const _Float16* br0 = vexp + ((size_t)e0 * RK + ln) * DIM + hb * 8;
  const _Float16* br1 = vexp + ((size_t)e1 * RK + ln) * DIM + hb * 8;

  v8f acc0 = vzero(), acc1 = vzero();
#pragma unroll 1
  for (int kk = 0; kk < DIM; kk += 32) {
    v16h a = ld_afrag(arow + kk + hb * 8);
    acc0 = wmma16(a, ld_bfrag(br0 + kk), acc0);
    acc1 = wmma16(a, ld_bfrag(br1 + kk), acc1);
  }
  float sc0 = sexp[e0 * RK + ln] * gates[b * 2 + 0];
  float sc1 = sexp[e1 * RK + ln] * gates[b * 2 + 1];
#pragma unroll
  for (int j = 0; j < 8; ++j) {
    int t = mt * 16 + hb * 8 + j;
    _Float16* dst = xvs + ((size_t)b * TP + t) * KR;
    dst[ln]      = (_Float16)(acc0[j] * sc0);
    dst[16 + ln] = (_Float16)(acc1[j] * sc1);
  }
}

// ---------------- main projection: Y = X@W^T + xv@Ucomb + bias ---------------
// Wave tile: 32(M) x 64(N).  All 8 waves of a block share the same N strip, so
// their W fragment loads are identical addresses (WGP-cache hits).
// MODE 0: f16 natural [NB][TP][DIM]   (Q, K)
// MODE 1: f16 transposed per head Vt[NB][NH][DH][SP]   (V)
// MODE 2: f32 final [NB][SEQ][DIM]    (O projection)

template <int MODE>
__global__ void k_proj(const _Float16* __restrict__ xin, const _Float16* __restrict__ wt,
                       const _Float16* __restrict__ xvs, const _Float16* __restrict__ uct,
                       const float* __restrict__ bias, float scale,
                       _Float16* __restrict__ outh, float* __restrict__ outf) {
  int w    = blockIdx.x * 8 + (threadIdx.x >> 5);
  int lane = threadIdx.x & 31;
  int ns = w / (NB * NT32);      // N strip of 64 (slowest: blocks share it)
  int mf = w % (NB * NT32);
  int b = mf / NT32, mt = mf % NT32;
  int ln = lane & 15, hb = lane >> 4;

  const _Float16* arow0 = xin + ((size_t)(b * TP + mt * 32 + ln)) * DIM;
  const _Float16* arow1 = arow0 + (size_t)16 * DIM;
  const _Float16* wrow[4];
#pragma unroll
  for (int nt = 0; nt < 4; ++nt) {
    int n = ns * 64 + nt * 16 + ln;
    wrow[nt] = wt + (size_t)n * DIM + hb * 8;
  }

  v8f acc[2][4];
#pragma unroll
  for (int i = 0; i < 2; ++i)
#pragma unroll
    for (int j = 0; j < 4; ++j) acc[i][j] = vzero();

#pragma unroll 1
  for (int kk = 0; kk < DIM; kk += 32) {
    __builtin_prefetch(arow0 + kk + 256, 0, 0);   // stream-ahead on A
    v16h a0 = ld_afrag(arow0 + kk + hb * 8);
    v16h a1 = ld_afrag(arow1 + kk + hb * 8);
#pragma unroll
    for (int nt = 0; nt < 4; ++nt) {
      v16h bb = ld_bfrag(wrow[nt] + kk);
      acc[0][nt] = wmma16(a0, bb, acc[0][nt]);
      acc[1][nt] = wmma16(a1, bb, acc[1][nt]);
    }
  }
  {  // fused MoE low-rank K=32 step
    v16h a0 = ld_afrag(xvs + ((size_t)(b * TP) + mt * 32 + ln) * KR + hb * 8);
    v16h a1 = ld_afrag(xvs + ((size_t)(b * TP) + mt * 32 + 16 + ln) * KR + hb * 8);
#pragma unroll
    for (int nt = 0; nt < 4; ++nt) {
      int n = ns * 64 + nt * 16 + ln;
      v16h bb = ld_bfrag(uct + ((size_t)b * DIM + n) * KR + hb * 8);
      acc[0][nt] = wmma16(a0, bb, acc[0][nt]);
      acc[1][nt] = wmma16(a1, bb, acc[1][nt]);
    }
  }

#pragma unroll
  for (int m2 = 0; m2 < 2; ++m2) {
    int rowbase = mt * 32 + m2 * 16;
#pragma unroll
    for (int nt = 0; nt < 4; ++nt) {
      int n = ns * 64 + nt * 16 + ln;
      float bv = bias[n];
      if (MODE == 0) {
#pragma unroll
        for (int j = 0; j < 8; ++j) {
          int t = rowbase + hb * 8 + j;
          outh[((size_t)(b * TP) + t) * DIM + n] =
              (_Float16)((acc[m2][nt][j] + bv) * scale);
        }
      } else if (MODE == 1) {
        H8 pk;
#pragma unroll
        for (int j = 0; j < 8; ++j)
          pk.h[j] = (_Float16)((acc[m2][nt][j] + bv) * scale);
        int h = n >> 6, d = n & 63;
        *(uint4*)(outh + (((size_t)(b * NH + h) * DH + d) * SP + rowbase + hb * 8)) =
            pk.u;
      } else {
#pragma unroll
        for (int j = 0; j < 8; ++j) {
          int t = rowbase + hb * 8 + j;
          if (t < SEQ)
            outf[((size_t)b * SEQ + t) * DIM + n] = (acc[m2][nt][j] + bv) * scale;
        }
      }
    }
  }
}

// ---------------- flash attention, one wave per (b, h, 16-row Q tile) --------
// Computes S^T = K @ Q^T so that exp'd score tiles are directly the A-fragment
// of P for the P@V WMMA (no cross-lane transpose needed).

__global__ void k_attn(const _Float16* __restrict__ q, const _Float16* __restrict__ kmat,
                       const _Float16* __restrict__ vt, _Float16* __restrict__ o) {
  int w    = blockIdx.x * 8 + (threadIdx.x >> 5);
  int lane = threadIdx.x & 31;
  int b = w / (NH * NT16);
  int rem = w % (NH * NT16);
  int h = rem / NT16, mt = rem % NT16;
  int ln = lane & 15, hb = lane >> 4;

  // Q^T B-fragments (reused across all key chunks)
  const _Float16* qrow = q + ((size_t)(b * TP) + mt * 16 + ln) * DIM + h * DH + hb * 8;
  v16h bq0 = ld_bfrag(qrow);
  v16h bq1 = ld_bfrag(qrow + 32);

  v8f oacc[4];
#pragma unroll
  for (int i = 0; i < 4; ++i) oacc[i] = vzero();
  float m_i = -1e30f, l_i = 0.f;

#pragma unroll 1
  for (int sc = 0; sc < 9; ++sc) {
    int s0 = sc * 32;
    // score tiles S^T (rows = keys, cols = queries); padded rows are in-bounds
    const _Float16* k0 = kmat + ((size_t)(b * TP) + s0 + ln) * DIM + h * DH + hb * 8;
    const _Float16* k1 = kmat + ((size_t)(b * TP) + s0 + 16 + ln) * DIM + h * DH + hb * 8;
    v8f c0 = vzero(), c1 = vzero();
    c0 = wmma16(ld_afrag(k0), bq0, c0);
    c0 = wmma16(ld_afrag(k0 + 32), bq1, c0);
    c1 = wmma16(ld_afrag(k1), bq0, c1);
    c1 = wmma16(ld_afrag(k1 + 32), bq1, c1);

    // mask invalid keys, running max
    float p0[8], p1[8];
    float mloc = -1e30f;
    int sb = s0 + hb * 8;
#pragma unroll
    for (int j = 0; j < 8; ++j) {
      float x0 = (sb + j < SEQ) ? c0[j] : -1e30f;
      float x1 = (sb + 16 + j < SEQ) ? c1[j] : -1e30f;
      p0[j] = x0; p1[j] = x1;
      mloc = fmaxf(mloc, fmaxf(x0, x1));
    }
    mloc = fmaxf(mloc, __shfl_xor(mloc, 16, 32));
    float mnew  = fmaxf(m_i, mloc);
    float alpha = __expf(m_i - mnew);
    m_i = mnew;

    float lsum = 0.f;
    v16h ap;  // A-fragment of P — identical layout to the C fragments!
#pragma unroll
    for (int j = 0; j < 8; ++j) {
      float e0 = __expf(p0[j] - mnew);
      float e1 = __expf(p1[j] - mnew);
      lsum += e0 + e1;
      ap[j]     = (_Float16)e0;
      ap[8 + j] = (_Float16)e1;
    }
    lsum += __shfl_xor(lsum, 16, 32);
    l_i = l_i * alpha + lsum;

    // rescale O accumulators (broadcast alpha of row j+hb*8 from its owner lane)
#pragma unroll
    for (int j = 0; j < 8; ++j) {
      float ab = __shfl(alpha, hb * 8 + j, 32);
      oacc[0][j] *= ab; oacc[1][j] *= ab; oacc[2][j] *= ab; oacc[3][j] *= ab;
    }
    // O += P @ V   (V pre-transposed per head: contiguous B-fragments)
#pragma unroll
    for (int dt = 0; dt < 4; ++dt) {
      int d = dt * 16 + ln;
      v16h bv = ld_bfrag(vt + (((size_t)(b * NH + h) * DH + d) * SP) + s0 + hb * 8);
      oacc[dt] = wmma16(ap, bv, oacc[dt]);
    }
  }

#pragma unroll
  for (int j = 0; j < 8; ++j) {
    float lb = __shfl(l_i, hb * 8 + j, 32);
    float r  = 1.f / lb;
    int t = mt * 16 + hb * 8 + j;
#pragma unroll
    for (int dt = 0; dt < 4; ++dt)
      o[((size_t)(b * TP) + t) * DIM + h * DH + dt * 16 + ln] =
          (_Float16)(oacc[dt][j] * r);
  }
}

// ---------------- host launch ----------------

extern "C" void kernel_launch(void* const* d_in, const int* in_sizes, int n_in,
                              void* d_out, int out_size, void* d_ws, size_t ws_size,
                              hipStream_t stream) {
  (void)in_sizes; (void)n_in; (void)out_size; (void)ws_size;
  const float* x     = (const float*)d_in[0];
  const float* gates = (const float*)d_in[1];
  const int*   idx   = (const int*)d_in[2];
  const float* wm[4], *um[4], *sm[4], *vm[4], *bm[4];
  for (int p = 0; p < 4; ++p) {
    wm[p] = (const float*)d_in[3 + 5 * p];
    um[p] = (const float*)d_in[4 + 5 * p];
    sm[p] = (const float*)d_in[5 + 5 * p];
    vm[p] = (const float*)d_in[6 + 5 * p];
    bm[p] = (const float*)d_in[7 + 5 * p];
  }
  float* out = (float*)d_out;

  char* ws = (char*)d_ws;
  auto alloc = [&](size_t bytes) {
    char* p = ws;
    ws += (bytes + 255) & ~(size_t)255;
    return p;
  };
  const size_t NXB = (size_t)NB * TP * DIM;       // activation elements
  _Float16* Xbf  = (_Float16*)alloc(NXB * 2);     // X (f16); later reused as attn output O
  _Float16* Qbf  = (_Float16*)alloc(NXB * 2);
  _Float16* Kbf  = (_Float16*)alloc(NXB * 2);
  _Float16* Vt   = (_Float16*)alloc((size_t)NB * NH * DH * SP * 2);
  _Float16* Wbf  = (_Float16*)alloc(4 * (size_t)DIM * DIM * 2);
  _Float16* Vexp = (_Float16*)alloc(4 * (size_t)NE * RK * DIM * 2);
  _Float16* Uct  = (_Float16*)alloc(4 * (size_t)NB * DIM * KR * 2);
  _Float16* Xvs  = (_Float16*)alloc(4 * (size_t)NB * TP * KR * 2);
  _Float16* Obf  = Xbf;  // alias: X no longer needed once attention runs

  const size_t NW  = (size_t)DIM * DIM;        // weight elems per projection
  const size_t NV  = (size_t)NE * RK * DIM;    // expert V elems per projection
  const size_t NU  = (size_t)NB * DIM * KR;    // Ucomb^T elems per projection
  const size_t NXV = (size_t)NB * TP * KR;     // xv elems per projection

  k_cvt_x<<<(unsigned)((NXB + 255) / 256), 256, 0, stream>>>(x, Xbf);
  for (int p = 0; p < 4; ++p) {
    k_cvt<<<(unsigned)((NW + 255) / 256), 256, 0, stream>>>(wm[p], Wbf + p * NW, (int)NW);
    k_cvt<<<(unsigned)((NV + 255) / 256), 256, 0, stream>>>(vm[p], Vexp + p * NV, (int)NV);
    k_ucombt<<<(NB * DIM) / 256, 256, 0, stream>>>(um[p], idx, Uct + p * NU);
  }

  const unsigned gXV = NB * NT16 / 8;             // 144 blocks (8 waves each)
  const unsigned gPR = 16 * NB * NT32 / 8;        // 1152
  const unsigned gAT = NB * NH * NT16 / 8;        // 2304

  // xv for q, k, v from X
  for (int p = 0; p < 3; ++p)
    k_xv<<<gXV, 256, 0, stream>>>(Xbf, Vexp + p * NV, sm[p], gates, idx, Xvs + p * NXV);

  // projections (ATT_SCALE folded into Q)
  k_proj<0><<<gPR, 256, 0, stream>>>(Xbf, Wbf + 0 * NW, Xvs + 0 * NXV, Uct + 0 * NU,
                                     bm[0], ATT_SCALE, Qbf, nullptr);
  k_proj<0><<<gPR, 256, 0, stream>>>(Xbf, Wbf + 1 * NW, Xvs + 1 * NXV, Uct + 1 * NU,
                                     bm[1], 1.f, Kbf, nullptr);
  k_proj<1><<<gPR, 256, 0, stream>>>(Xbf, Wbf + 2 * NW, Xvs + 2 * NXV, Uct + 2 * NU,
                                     bm[2], 1.f, Vt, nullptr);

  // attention (writes Obf == Xbf; X fully consumed above)
  k_attn<<<gAT, 256, 0, stream>>>(Qbf, Kbf, Vt, Obf);

  // MoE xv for the output projection, then final projection to fp32 d_out
  k_xv<<<gXV, 256, 0, stream>>>(Obf, Vexp + 3 * NV, sm[3], gates, idx, Xvs + 3 * NXV);
  k_proj<2><<<gPR, 256, 0, stream>>>(Obf, Wbf + 3 * NW, Xvs + 3 * NXV, Uct + 3 * NU,
                                     bm[3], 1.f, nullptr, out);
}
